// PG_few_63831803953155
// MI455X (gfx1250) — matrix operand order repaired and checked by reading, not compiled, not run on hardware
//
#include <hip/hip_runtime.h>
#include <hip/hip_bf16.h>
#include <math.h>

// ---------- CDNA5 WMMA types ----------
typedef __attribute__((ext_vector_type(16))) __bf16 v16bf;
typedef __attribute__((ext_vector_type(8)))  float  v8f;

union V16 {
    v16bf v;
    uint4 q[2];
    unsigned short u[16];
};
union Acc8 {
    v8f   v;
    float f[8];
};

// ---------- problem constants (reference: N=100000, D=256, B=1024, K=128) ----------
#define DD     256          // feature dim
#define KK     128          // nodes per ego-graph
#define TBS    264          // LDS row stride in bf16 elements (256 + 8 pad -> 528B, 16B aligned)
#define SIMS   132          // LDS sim row stride in floats (128 + 4 pad -> 528B)

#define NODES_OFF   0                       // 128 ints            = 512 B
#define TBH_OFF     512                     // 128*264 shorts      = 67584 B
#define TBL_OFF     (512 + 67584)           // 128*264 shorts      = 67584 B
#define SIM_OFF     (512 + 2*67584)         // 128*132 floats      = 67584 B
#define RED_OFF     (512 + 3*67584)         // 16 floats
#define ROWV_OFF    (RED_OFF + 64)          // 128 floats
#define SCAL_OFF    (ROWV_OFF + 512)        // 4 floats
#define SMEM_BYTES  (SCAL_OFF + 16)         // ~203,872 B (< 320KB WGP LDS)

// ---------- reductions ----------
__device__ __forceinline__ float block_sum(float v, float* buf) {
    const int lane = threadIdx.x & 31;
    const int w    = threadIdx.x >> 5;
    #pragma unroll
    for (int m = 16; m > 0; m >>= 1) v += __shfl_xor(v, m, 32);
    __syncthreads();                 // protect buf reuse across calls
    if (lane == 0) buf[w] = v;
    __syncthreads();
    float r = 0.0f;
    const int nw = blockDim.x >> 5;
    for (int i = 0; i < nw; ++i) r += buf[i];
    return r;
}

__device__ __forceinline__ unsigned short f32_to_bf16_rne(float x) {
    unsigned u = __float_as_uint(x);
    unsigned r = u + 0x7FFFu + ((u >> 16) & 1u);
    return (unsigned short)(r >> 16);
}

// =====================================================================
// Kernel A: per-row inverse norms: inv_f, inv_e, inv_te   (grid = N)
// =====================================================================
__global__ __launch_bounds__(256)
void norms_kernel(const float* __restrict__ feature,
                  const float* __restrict__ temb,
                  const float* __restrict__ virt,
                  float* __restrict__ invf,
                  float* __restrict__ inve,
                  float* __restrict__ invte) {
    __shared__ float buf[8];
    const int n = blockIdx.x;
    const int d = threadIdx.x;          // blockDim == DD == 256
    const size_t base = (size_t)n * DD + d;

    const float f = feature[base];
    const float e = temb[base];

    const float ss_f = block_sum(f * f, buf);
    const float ss_e = block_sum(e * e, buf);
    const float ie   = 1.0f / fmaxf(sqrtf(ss_e), 1e-12f);

    const float t    = e * ie - virt[d];
    const float ss_t = block_sum(t * t, buf);

    if (d == 0) {
        invf[n]  = 1.0f / fmaxf(sqrtf(ss_f), 1e-12f);
        inve[n]  = ie;
        invte[n] = 1.0f / fmaxf(sqrtf(ss_t), 1e-12f);
    }
}

// =====================================================================
// Kernel B: per-batch: gather+bf16split -> WMMA Gram -> softmax -> score
// grid = B, 256 threads (8 waves), dynamic LDS = SMEM_BYTES
// =====================================================================
__global__ __launch_bounds__(256)
void batch_kernel(const float* __restrict__ feature,
                  const float* __restrict__ temb,
                  const float* __restrict__ virt,
                  const float* __restrict__ gumbel,
                  const int*   __restrict__ ego,
                  const int*   __restrict__ adj,
                  const float* __restrict__ invf,
                  const float* __restrict__ inve,
                  const float* __restrict__ invte,
                  float* __restrict__ raw_score) {
    extern __shared__ unsigned char smem[];
    int*            nodes = (int*)(smem + NODES_OFF);
    unsigned short* tbh   = (unsigned short*)(smem + TBH_OFF);
    unsigned short* tbl   = (unsigned short*)(smem + TBL_OFF);
    float*          simr  = (float*)(smem + SIM_OFF);
    float*          red   = (float*)(smem + RED_OFF);
    float*          rowv  = (float*)(smem + ROWV_OFF);
    float*          scal  = (float*)(smem + SCAL_OFF);

    const int b   = blockIdx.x;
    const int tid = threadIdx.x;

    if (tid < KK) nodes[tid] = ego[(size_t)b * KK + tid];
    __syncthreads();

    // ---- Phase 1: gather rows, normalize, bf16 hi/lo split; fuse mean accumulation ----
    const float vd = virt[tid];         // d == tid (blockDim == DD)
    float sumF = 0.0f, sumE = 0.0f;
    for (int k = 0; k < KK; ++k) {
        const int node = nodes[k];
        const float ie  = inve[node];
        const float it  = invte[node];
        const float ifn = invf[node];
        const float e = temb[(size_t)node * DD + tid];
        const float f = feature[(size_t)node * DD + tid];
        sumE += e * ie;
        sumF += f * ifn;
        const float t  = (e * ie - vd) * it;
        const unsigned short hs = f32_to_bf16_rne(t);
        const float hf = __uint_as_float((unsigned)hs << 16);
        const unsigned short ls = f32_to_bf16_rne(t - hf);
        tbh[k * TBS + tid] = hs;
        tbl[k * TBS + tid] = ls;
    }

    // f_diff = || mean(feat) - mean(emb) ||
    const float dmean = (sumF - sumE) * (1.0f / (float)KK);
    const float fd2   = block_sum(dmean * dmean, red);   // includes __syncthreads (tb visible)
    if (tid == 0) scal[0] = sqrtf(fd2);
    __syncthreads();

    // ---- Phase 2: sim = T T^T via v_wmma_f32_16x16x32_bf16, bf16x2 (hi*hi + hi*lo + lo*hi) ----
    {
        const int w    = tid >> 5;            // wave id = m-tile (8 waves, 8 m-tiles)
        const int lane = tid & 31;
        const int hl   = lane >> 4;           // 0: lanes 0-15, 1: lanes 16-31
        const int l15  = lane & 15;
        const int aRow = w * 16 + l15;        // A row (M)
        const int aK0  = hl ? 8 : 0;          // A: K {0..7,16..23} vs {8..15,24..31}
        const int bK0  = hl ? 8 : 0;          // B: K {0..7,16..23} vs {8..15,24..31}

        for (int nt = 0; nt < 8; ++nt) {
            const int bRow = nt * 16 + l15;   // B column (N) comes from row bRow of T
            Acc8 acc;
            #pragma unroll
            for (int r = 0; r < 8; ++r) acc.f[r] = 0.0f;

            #pragma unroll
            for (int ks = 0; ks < 8; ++ks) {
                const int kb = ks * 32;
                V16 ah, al, bh, bl;
                const unsigned short* pa = tbh + aRow * TBS + kb + aK0;
                const unsigned short* qa = tbl + aRow * TBS + kb + aK0;
                ah.q[0] = *(const uint4*)(pa);       // K +0..7
                ah.q[1] = *(const uint4*)(pa + 16);  // K +16..23
                al.q[0] = *(const uint4*)(qa);
                al.q[1] = *(const uint4*)(qa + 16);
                const unsigned short* pb = tbh + bRow * TBS + kb + bK0;
                const unsigned short* qb = tbl + bRow * TBS + kb + bK0;
                bh.q[0] = *(const uint4*)(pb);       // K +0..7
                bh.q[1] = *(const uint4*)(pb + 16);  // K +16..23
                bl.q[0] = *(const uint4*)(qb);
                bl.q[1] = *(const uint4*)(qb + 16);

                acc.v = __builtin_amdgcn_wmma_f32_16x16x32_bf16(
                            false, ah.v, false, bh.v, (short)0, acc.v, false, false);
                acc.v = __builtin_amdgcn_wmma_f32_16x16x32_bf16(
                            false, ah.v, false, bl.v, (short)0, acc.v, false, false);
                acc.v = __builtin_amdgcn_wmma_f32_16x16x32_bf16(
                            false, al.v, false, bh.v, (short)0, acc.v, false, false);
            }
            // C/D layout: VGPR r -> M = 16w + (hl?8:0) + r ; lane -> N = 16nt + l15
            const int col   = nt * 16 + l15;
            const int rbase = w * 16 + (hl ? 8 : 0);
            #pragma unroll
            for (int r = 0; r < 8; ++r)
                simr[(rbase + r) * SIMS + col] = acc.f[r];
        }
    }
    __syncthreads();

    // ---- Phase 3: row-wise gumbel-softmax + ||adj - p||_2 ----
    if (tid < KK) {
        const float* g = gumbel + ((size_t)b * KK + tid) * KK;
        const int*   a = adj    + ((size_t)b * KK + tid) * KK;
        float* s = simr + tid * SIMS;
        float m = -3.0e38f;
        for (int j = 0; j < KK; ++j) {
            const float v = s[j] - 0.5f + g[j];   // sim - THRESHOLD + gumbel
            s[j] = v;
            m = fmaxf(m, v);
        }
        float sum = 0.0f;
        for (int j = 0; j < KK; ++j) sum += __expf(s[j] - m);
        const float inv = 1.0f / sum;
        float accd = 0.0f;
        for (int j = 0; j < KK; ++j) {
            const float p  = __expf(s[j] - m) * inv;
            const float dd = (float)a[j] - p;
            accd += dd * dd;
        }
        rowv[tid] = sqrtf(accd);
    }
    __syncthreads();

    const float sd = block_sum(tid < KK ? rowv[tid] : 0.0f, red);
    if (tid == 0) raw_score[b] = sd * (1.0f / (float)KK) + scal[0];
}

// =====================================================================
// Kernel C: global min/max normalize + BCE loss  (single block)
// =====================================================================
__global__ __launch_bounds__(256)
void finalize_kernel(const float* __restrict__ raw,
                     const int*   __restrict__ label,
                     float* __restrict__ out, int B) {
    __shared__ float bmin[8], bmax[8], bsum[8];
    const int tid  = threadIdx.x;
    const int lane = tid & 31;
    const int w    = tid >> 5;

    float mn = 3.0e38f, mx = -3.0e38f;
    for (int i = tid; i < B; i += blockDim.x) {
        const float v = raw[i];
        mn = fminf(mn, v);
        mx = fmaxf(mx, v);
    }
    #pragma unroll
    for (int m = 16; m > 0; m >>= 1) {
        mn = fminf(mn, __shfl_xor(mn, m, 32));
        mx = fmaxf(mx, __shfl_xor(mx, m, 32));
    }
    if (lane == 0) { bmin[w] = mn; bmax[w] = mx; }
    __syncthreads();
    mn = bmin[0]; mx = bmax[0];
    for (int i = 1; i < (int)(blockDim.x >> 5); ++i) {
        mn = fminf(mn, bmin[i]);
        mx = fmaxf(mx, bmax[i]);
    }
    const float scale = 1.0f / (mx - mn);

    float lsum = 0.0f;
    for (int i = tid; i < B; i += blockDim.x) {
        const float sc = (raw[i] - mn) * scale;
        out[i] = sc;
        const float scc = fminf(fmaxf(sc, 1e-7f), 1.0f - 1e-7f);
        const float y   = (float)label[i];
        lsum += -(y * logf(scc) + (1.0f - y) * log1pf(-scc));
    }
    #pragma unroll
    for (int m = 16; m > 0; m >>= 1) lsum += __shfl_xor(lsum, m, 32);
    if (lane == 0) bsum[w] = lsum;
    __syncthreads();
    if (tid == 0) {
        float tot = 0.0f;
        for (int i = 0; i < (int)(blockDim.x >> 5); ++i) tot += bsum[i];
        out[B] = tot / (float)B / (float)B;   // bce.mean() / B
    }
}

// =====================================================================
extern "C" void kernel_launch(void* const* d_in, const int* in_sizes, int n_in,
                              void* d_out, int out_size, void* d_ws, size_t ws_size,
                              hipStream_t stream) {
    const float* feature = (const float*)d_in[0];
    const float* temb    = (const float*)d_in[1];
    const float* virt    = (const float*)d_in[2];
    const float* gumbel  = (const float*)d_in[3];
    const int*   ego     = (const int*)d_in[4];
    const int*   adj     = (const int*)d_in[5];
    const int*   label   = (const int*)d_in[6];

    const int N = in_sizes[0] / DD;     // 100000
    const int B = in_sizes[6];          // 1024

    float* ws    = (float*)d_ws;
    float* invf  = ws;
    float* inve  = ws + (size_t)N;
    float* invte = ws + (size_t)2 * N;
    float* raw   = ws + (size_t)3 * N;

    (void)hipFuncSetAttribute((const void*)batch_kernel,
                              hipFuncAttributeMaxDynamicSharedMemorySize,
                              SMEM_BYTES);

    norms_kernel<<<N, 256, 0, stream>>>(feature, temb, virt, invf, inve, invte);
    batch_kernel<<<B, 256, SMEM_BYTES, stream>>>(feature, temb, virt, gumbel,
                                                 ego, adj, invf, inve, invte, raw);
    finalize_kernel<<<1, 256, 0, stream>>>(raw, label, (float*)d_out, B);
}
